// SASS2D_5179730559105
// MI455X (gfx1250) — matrix operand order repaired and checked by reading, not compiled, not run on hardware
//
#include <hip/hip_runtime.h>
#include <hip/hip_bf16.h>

typedef __attribute__((ext_vector_type(16))) _Float16 v16h;
typedef __attribute__((ext_vector_type(8)))  _Float16 v8h;
typedef __attribute__((ext_vector_type(4)))  _Float16 v4h;
typedef __attribute__((ext_vector_type(8)))  float    v8f;

#define C_DIM 256
#define G_DIM 16
#define H_DIM 80
#define W_DIM 80
#define P_DIM 6400          // H*W
#define CP_DIM 1638400      // C*P
#define N_IMG 8
#define TOTAL_ELEMS 13107200
#define GN_COUNT 102400.0f  // (C/G)*H*W
#define EPS_GN 1e-5f
#define XS_STRIDE 40        // LDS halves per pixel row (32 K + 8 pad -> bank-spread, 16B aligned)

// ---------------------------------------------------------------------------
// K0: convert the three 1x1-conv weight matrices to f16, zero GN stat buffers
// ---------------------------------------------------------------------------
__global__ void prep_kernel(const float* __restrict__ wpre,
                            const float* __restrict__ wgate,
                            const float* __restrict__ wpw,
                            _Float16* __restrict__ fpre,
                            _Float16* __restrict__ fgate,
                            _Float16* __restrict__ fpw,
                            float* __restrict__ stats1,
                            float* __restrict__ stats2) {
    int i = blockIdx.x * blockDim.x + threadIdx.x;   // 65536 threads
    if (i < C_DIM * C_DIM) {
        fpre[i]  = (_Float16)wpre[i];
        fgate[i] = (_Float16)wgate[i];
        fpw[i]   = (_Float16)wpw[i];
    }
    if (i < 2 * N_IMG * G_DIM) {                     // 256 floats each
        stats1[i] = 0.0f;
        stats2[i] = 0.0f;
    }
}

// ---------------------------------------------------------------------------
// WMMA f16 GEMM:  Y[c, pix] = sum_k W[c,k] * X[k, pix]   (per image, NCHW)
//   block = 256 threads = 8 waves; block tile = ALL 256 channels x 64 pixels
//   wave tile = 32 channels x 64 pixels  (2 M-tiles x 4 N-tiles -> 8 WMMA/K-step)
//   X K-slice staged once per block through double-buffered LDS in the exact
//   WMMA B layout ([pixel][K] f16) -> B frag = 2x ds_load_b128 per tile.
// mode 0/2: store raw Y + atomic GroupNorm sum/sumsq per (n, group)
// mode 1  : gated epilogue: Y = 0.25*(RS+CS+2h) * sigmoid(acc + b_gate)
// ---------------------------------------------------------------------------
__global__ __launch_bounds__(256)
void gemm_wmma(const _Float16* __restrict__ Wf16,
               const float*    __restrict__ X,
               float*          __restrict__ Y,
               float*          __restrict__ stats,
               const float*    __restrict__ Hfeat,
               const float*    __restrict__ RS,
               const float*    __restrict__ CS,
               const float*    __restrict__ bgate,
               int mode) {
    __shared__ _Float16 xs[2][64 * XS_STRIDE];   // 2 x 5120 B staging buffers

    const int tid  = threadIdx.x;
    const int wave = tid >> 5;
    const int lane = tid & 31;
    const int lo   = lane & 15;     // N index / A row index
    const int hi   = lane >> 4;     // K-half selector

    const int q_base = blockIdx.x * 64;          // 64-pixel tile (one image)
    const int n_img  = q_base / P_DIM;
    const int p_base = q_base - n_img * P_DIM;
    const int c_wave = wave * 32;                // 32 channels per wave

    const float* Xn = X + (size_t)n_img * CP_DIM;

    // staging role: this thread fills 2 pixels x 4 K values of the LDS tile
    const int p2 = lane * 2;                     // local pixel pair
    const int kq = wave * 4;                     // K quad within the 32-K slice

    float2 rx[4];
    auto gload = [&](int ks) {
        const float* src = Xn + (size_t)(ks * 32 + kq) * P_DIM + (p_base + p2);
#pragma unroll
        for (int i = 0; i < 4; i++)
            rx[i] = *(const float2*)(src + (size_t)i * P_DIM);   // coalesced 256B/instr
    };
    auto lstore = [&](int b) {
        v4h h0, h1;
#pragma unroll
        for (int i = 0; i < 4; i++) {
            h0[i] = (_Float16)rx[i].x;
            h1[i] = (_Float16)rx[i].y;
        }
        *(v4h*)&xs[b][p2 * XS_STRIDE + kq]       = h0;
        *(v4h*)&xs[b][(p2 + 1) * XS_STRIDE + kq] = h1;
    };

    v8f acc[2][4] = {};

    gload(0);
    lstore(0);
    gload(1);
    __syncthreads();

    for (int ks = 0; ks < 8; ks++) {             // K = ks*32 .. +31
        const int cur = ks & 1;
        if (ks < 7) lstore(cur ^ 1);             // tile ks+1 -> other buffer
        if (ks < 6) gload(ks + 2);               // prefetch tile ks+2

        const int k0 = ks * 32;

        // A fragments: W rows, K chunk (reused across the 4 N-tiles)
        v16h A[2];
#pragma unroll
        for (int mt = 0; mt < 2; mt++) {
            const _Float16* wrow =
                Wf16 + (size_t)(c_wave + mt * 16 + lo) * C_DIM + k0 + hi * 8;
            v8h a0 = *(const v8h*)(wrow);        // VGPR0-3: K = hi*8 + 0..7
            v8h a1 = *(const v8h*)(wrow + 16);   // VGPR4-7: K = 16 + hi*8 + 0..7
#pragma unroll
            for (int i = 0; i < 8; i++) { A[mt][i] = a0[i]; A[mt][i + 8] = a1[i]; }
        }

#pragma unroll
        for (int t = 0; t < 4; t++) {
            // B fragment straight out of LDS: K = hi*8 + 0..7 and 16 + hi*8 + 0..7
            const _Float16* bp = &xs[cur][(t * 16 + lo) * XS_STRIDE + hi * 8];
            v8h b0 = *(const v8h*)(bp);
            v8h b1 = *(const v8h*)(bp + 16);
            v16h B;
#pragma unroll
            for (int i = 0; i < 8; i++) { B[i] = b0[i]; B[i + 8] = b1[i]; }
#pragma unroll
            for (int mt = 0; mt < 2; mt++)
                acc[mt][t] = __builtin_amdgcn_wmma_f32_16x16x32_f16(
                    false, A[mt], false, B, (short)0, acc[mt][t], false, false);
        }
        __syncthreads();
    }

    if (mode != 1) {
        // raw store + fused GroupNorm partial stats (each M-tile = one group)
#pragma unroll
        for (int mt = 0; mt < 2; mt++) {
            float s = 0.0f, ss = 0.0f;
#pragma unroll
            for (int t = 0; t < 4; t++) {
                const int p = p_base + t * 16 + lo;
#pragma unroll
                for (int r = 0; r < 8; r++) {
                    const int c = c_wave + mt * 16 + r + hi * 8;
                    float v = acc[mt][t][r];
                    Y[(size_t)n_img * CP_DIM + (size_t)c * P_DIM + p] = v;
                    s += v; ss += v * v;
                }
            }
#pragma unroll
            for (int m = 16; m > 0; m >>= 1) {
                s  += __shfl_xor(s,  m, 32);
                ss += __shfl_xor(ss, m, 32);
            }
            if (lane == 0) {
                const int sidx = (n_img * G_DIM + (wave * 2 + mt)) * 2;
                atomicAdd(&stats[sidx],     s);
                atomicAdd(&stats[sidx + 1], ss);
            }
        }
    } else {
        // gate epilogue: fused scans collapse to RS + CS + 2h
#pragma unroll
        for (int mt = 0; mt < 2; mt++) {
#pragma unroll
            for (int t = 0; t < 4; t++) {
                const int p = p_base + t * 16 + lo;
                const int i = p / W_DIM;
                const int j = p - i * W_DIM;
#pragma unroll
                for (int r = 0; r < 8; r++) {
                    const int c = c_wave + mt * 16 + r + hi * 8;
                    const size_t idx =
                        (size_t)n_img * CP_DIM + (size_t)c * P_DIM + p;
                    const float hv = Hfeat[idx];
                    const float y  = acc[mt][t][r] + bgate[c];
                    const float gt = 1.0f / (1.0f + __expf(-y));
                    const float fused =
                        0.25f * (RS[(size_t)(n_img * C_DIM + c) * H_DIM + i]
                               + CS[(size_t)(n_img * C_DIM + c) * W_DIM + j]
                               + 2.0f * hv);
                    Y[idx] = fused * gt;
                }
            }
        }
    }
}

// ---------------------------------------------------------------------------
// K2: h = silu(groupnorm(y1));  fused row-sums RS[n,c,i] and col-sums CS[n,c,j]
//     one workgroup per (n,c) plane of 6400 elements
// ---------------------------------------------------------------------------
__global__ __launch_bounds__(256)
void gn_silu_rowcol(const float* __restrict__ Y1,
                    const float* __restrict__ stats,
                    const float* __restrict__ g1,
                    const float* __restrict__ b1,
                    float* __restrict__ Hout,
                    float* __restrict__ RS,
                    float* __restrict__ CS) {
    __shared__ float rs[H_DIM];
    __shared__ float cs[W_DIM];

    const int nc = blockIdx.x;          // 0..2047
    const int n  = nc >> 8;
    const int c  = nc & 255;
    const int g  = c >> 4;

    const float s   = stats[(n * G_DIM + g) * 2];
    const float ssq = stats[(n * G_DIM + g) * 2 + 1];
    const float inv = 1.0f / GN_COUNT;
    const float mu  = s * inv;
    const float rstd = rsqrtf(ssq * inv - mu * mu + EPS_GN);
    const float ga = g1[c] * rstd;
    const float be = b1[c] - mu * ga;

    const int t = threadIdx.x;
    if (t < H_DIM) { rs[t] = 0.0f; cs[t] = 0.0f; }
    __syncthreads();

    const float* src = Y1   + (size_t)nc * P_DIM;
    float*       dst = Hout + (size_t)nc * P_DIM;
#pragma unroll
    for (int k = 0; k < 25; k++) {
        const int p = t + k * 256;
        const float v  = src[p] * ga + be;
        const float hv = v / (1.0f + __expf(-v));   // silu
        dst[p] = hv;
        const int i = p / W_DIM;
        const int j = p - i * W_DIM;
        atomicAdd(&rs[i], hv);
        atomicAdd(&cs[j], hv);
    }
    __syncthreads();
    if (t < H_DIM) {
        RS[(size_t)nc * H_DIM + t] = rs[t];
        CS[(size_t)nc * W_DIM + t] = cs[t];
    }
}

// ---------------------------------------------------------------------------
// K4: depthwise 3x3 (cross-correlation, pad 1), plane staged in LDS
// ---------------------------------------------------------------------------
__global__ __launch_bounds__(256)
void dwconv3x3(const float* __restrict__ Gt,
               const float* __restrict__ Wdw,
               float* __restrict__ Z) {
    __shared__ float tile[P_DIM];       // 25.6 KB of the 320 KB LDS

    const int nc = blockIdx.x;
    const int c  = nc & 255;
    const int t  = threadIdx.x;

    const float* src = Gt + (size_t)nc * P_DIM;
#pragma unroll
    for (int k = 0; k < 25; k++) tile[t + k * 256] = src[t + k * 256];

    float w[9];
#pragma unroll
    for (int k = 0; k < 9; k++) w[k] = Wdw[c * 9 + k];
    __syncthreads();

    float* dst = Z + (size_t)nc * P_DIM;
#pragma unroll
    for (int k = 0; k < 25; k++) {
        const int p = t + k * 256;
        const int i = p / W_DIM;
        const int j = p - i * W_DIM;
        float acc = 0.0f;
#pragma unroll
        for (int di = -1; di <= 1; di++) {
            const int ii = i + di;
            if (ii < 0 || ii >= H_DIM) continue;
#pragma unroll
            for (int dj = -1; dj <= 1; dj++) {
                const int jj = j + dj;
                if (jj < 0 || jj >= W_DIM) continue;
                acc += w[(di + 1) * 3 + (dj + 1)] * tile[ii * W_DIM + jj];
            }
        }
        dst[p] = acc;
    }
}

// ---------------------------------------------------------------------------
// K6: out = silu(groupnorm2(y2)), float4 vectorized
// ---------------------------------------------------------------------------
__global__ __launch_bounds__(256)
void gn_silu_out(const float* __restrict__ Y2,
                 const float* __restrict__ stats,
                 const float* __restrict__ g2,
                 const float* __restrict__ b2,
                 float* __restrict__ Out) {
    const size_t idx = ((size_t)blockIdx.x * 256 + threadIdx.x) * 4;  // grid=12800 exact
    const int n = (int)(idx / CP_DIM);
    const int c = (int)((idx / P_DIM) & 255);   // 4 consecutive elems share c (P%4==0)
    const int g = c >> 4;

    const float s   = stats[(n * G_DIM + g) * 2];
    const float ssq = stats[(n * G_DIM + g) * 2 + 1];
    const float inv = 1.0f / GN_COUNT;
    const float mu  = s * inv;
    const float rstd = rsqrtf(ssq * inv - mu * mu + EPS_GN);
    const float ga = g2[c] * rstd;
    const float be = b2[c] - mu * ga;

    float4 v = *(const float4*)(Y2 + idx);
    float4 o;
    o.x = (v.x * ga + be); o.x = o.x / (1.0f + __expf(-o.x));
    o.y = (v.y * ga + be); o.y = o.y / (1.0f + __expf(-o.y));
    o.z = (v.z * ga + be); o.z = o.z / (1.0f + __expf(-o.z));
    o.w = (v.w * ga + be); o.w = o.w / (1.0f + __expf(-o.w));
    *(float4*)(Out + idx) = o;
}

// ---------------------------------------------------------------------------
extern "C" void kernel_launch(void* const* d_in, const int* in_sizes, int n_in,
                              void* d_out, int out_size, void* d_ws, size_t ws_size,
                              hipStream_t stream) {
    const float* x      = (const float*)d_in[0];
    const float* w_pre  = (const float*)d_in[1];
    const float* g1     = (const float*)d_in[2];
    const float* b1     = (const float*)d_in[3];
    const float* w_gate = (const float*)d_in[4];
    const float* b_gate = (const float*)d_in[5];
    const float* w_dw   = (const float*)d_in[6];
    const float* w_pw   = (const float*)d_in[7];
    const float* g2     = (const float*)d_in[8];
    const float* b2     = (const float*)d_in[9];
    float* out = (float*)d_out;

    // workspace layout (bytes), ~107 MB total, all offsets 256B aligned
    char* ws = (char*)d_ws;
    float*     bufA   = (float*)(ws);                     // y1 -> gated   (52,428,800)
    float*     bufB   = (float*)(ws + 52428800);          // h  -> y2      (52,428,800)
    _Float16*  fpre   = (_Float16*)(ws + 104857600);      // 131,072
    _Float16*  fgate  = (_Float16*)(ws + 104988672);      // 131,072
    _Float16*  fpw    = (_Float16*)(ws + 105119744);      // 131,072
    float*     stats1 = (float*)(ws + 105250816);         // 1,024
    float*     stats2 = (float*)(ws + 105251840);         // 1,024
    float*     RS     = (float*)(ws + 105252864);         // 655,360
    float*     CS     = (float*)(ws + 105908224);         // 655,360

    const int gemm_blocks = TOTAL_ELEMS / C_DIM / 64;     // 800 pixel tiles

    // K0: weight f16 conversion + zero stats
    prep_kernel<<<256, 256, 0, stream>>>(w_pre, w_gate, w_pw,
                                         fpre, fgate, fpw, stats1, stats2);
    // K1: y1 = w_pre @ x   (+ GN1 stats)
    gemm_wmma<<<gemm_blocks, 256, 0, stream>>>(fpre, x, bufA, stats1,
                                               nullptr, nullptr, nullptr, nullptr, 0);
    // K2: h = silu(gn1(y1)); RS/CS sums (scans collapsed algebraically)
    gn_silu_rowcol<<<N_IMG * C_DIM, 256, 0, stream>>>(bufA, stats1, g1, b1,
                                                      bufB, RS, CS);
    // K3: gated = 0.25*(RS+CS+2h) * sigmoid(w_gate @ h + b_gate)
    gemm_wmma<<<gemm_blocks, 256, 0, stream>>>(fgate, bufB, bufA, nullptr,
                                               bufB, RS, CS, b_gate, 1);
    // K4: depthwise 3x3 -> d_out used as scratch for z
    dwconv3x3<<<N_IMG * C_DIM, 256, 0, stream>>>(bufA, w_dw, out);
    // K5: y2 = w_pw @ z   (+ GN2 stats)
    gemm_wmma<<<gemm_blocks, 256, 0, stream>>>(fpw, out, bufB, stats2,
                                               nullptr, nullptr, nullptr, nullptr, 2);
    // K6: out = silu(gn2(y2))
    gn_silu_out<<<TOTAL_ELEMS / 1024, 256, 0, stream>>>(bufB, stats2, g2, b2, out);
}
